// GraphNetBlock_58514634441263
// MI455X (gfx1250) — matrix operand order repaired and checked by reading, not compiled, not run on hardware
//
#include <hip/hip_runtime.h>
#include <hip/hip_bf16.h>
#include <math.h>

typedef __attribute__((ext_vector_type(16))) __bf16        v16bf;
typedef __attribute__((ext_vector_type(8)))  float         v8f;
typedef __attribute__((ext_vector_type(4)))  unsigned int  u32x4;

#define DIM      128
#define N_NODES  50000
#define N_EDGES  600000
#define LN_EPS   1e-5f

// ---------- helpers ----------

static __device__ __forceinline__ unsigned short f2bf(float f) {
    unsigned int x = __float_as_uint(f);
    x += 0x7FFFu + ((x >> 16) & 1u);          // round-to-nearest-even
    return (unsigned short)(x >> 16);
}

static __device__ __forceinline__ void atomicMaxFloat(float* addr, float val) {
    if (val >= 0.0f) atomicMax((int*)addr, __float_as_int(val));
    else             atomicMin((unsigned int*)addr, __float_as_uint(val));
}

// A-fragment (16x32 bf16, ISA 7.12.2 layout) from LDS row-major bf16 tile.
// lane 0-15: m=lane,  K = kbase+0..7  and kbase+16..23
// lane16-31: m=lane-16, K = kbase+8..15 and kbase+24..31  (kb = 8)
static __device__ __forceinline__ v16bf load_a(const unsigned short* lds, int stride,
                                               int m, int kbase, int kb) {
    union { u32x4 q[2]; v16bf v; } f;
    f.q[0] = *(const u32x4*)(lds + m * stride + kbase + kb);
    f.q[1] = *(const u32x4*)(lds + m * stride + kbase + 16 + kb);
    return f.v;
}

// B-fragment from pre-packed weights: contiguous 32B per lane.
static __device__ __forceinline__ v16bf load_b(const unsigned int* packed,
                                               int kt, int nt, int lane) {
    union { u32x4 q[2]; v16bf v; } f;
    const unsigned int* p = packed + (size_t)(((kt * 8 + nt) * 32 + lane) * 8);
    f.q[0] = *(const u32x4*)(p);
    f.q[1] = *(const u32x4*)(p + 4);
    return f.v;
}

// ---------- prep kernels ----------

__global__ void init_kernel(float* segmax, float* segsum, float* agg) {
    unsigned int idx = blockIdx.x * 256u + threadIdx.x;
    if (idx < N_NODES) { segmax[idx] = __uint_as_float(0xFF800000u); segsum[idx] = 0.0f; }
    if (idx < (unsigned)N_NODES * DIM) agg[idx] = 0.0f;
}

// Pack W (Ktot x 128, row-major fp32) into bf16 B-fragment layout.
// dword p = [kt][nt][lane][j]: n = nt*16 + (lane&15), k = kt*32 + (lane>>4)*8 + (j>>2)*8 + 2j
__global__ void pack_weights(const float* __restrict__ W, unsigned int* __restrict__ dst, int KT) {
    int p = blockIdx.x * blockDim.x + threadIdx.x;
    if (p >= KT * 2048) return;
    int j    = p & 7;
    int lane = (p >> 3) & 31;
    int nt   = (p >> 8) & 7;
    int kt   = p >> 11;
    int n = nt * 16 + (lane & 15);
    int k = kt * 32 + ((lane >> 4) * 8) + ((j >> 2) * 8) + 2 * j;
    unsigned int lo = f2bf(W[(size_t)k * DIM + n]);
    unsigned int hi = f2bf(W[(size_t)(k + 1) * DIM + n]);
    dst[p] = lo | (hi << 16);
}

// ---------- edge MLP + LN + attention logits ----------

__global__ void __launch_bounds__(256)
edge_kernel(const float* __restrict__ nf, const float* __restrict__ ef,
            const int* __restrict__ senders, const int* __restrict__ receivers,
            const unsigned int* __restrict__ eW1p, const float* __restrict__ eb1,
            const unsigned int* __restrict__ eW2p, const float* __restrict__ eb2,
            const float* __restrict__ eg, const float* __restrict__ ebeta,
            const float* __restrict__ aW, const float* __restrict__ ab,
            float* __restrict__ out_ef, float* __restrict__ logits,
            float* __restrict__ segmax)
{
    __shared__ __align__(16) unsigned short x_lds[16 * 384];
    __shared__ __align__(16) unsigned short h_lds[16 * 128];
    __shared__ __align__(16) float          y_lds[16 * 128];
    __shared__ float red_s[256];
    __shared__ float red_q[256];
    __shared__ float mu_lds[16], rs_lds[16];
    __shared__ int   s_idx[16], r_idx[16];

    const int tid  = threadIdx.x;
    const int lane = tid & 31;
    const int wv   = tid >> 5;                    // N-tile 0..7
    const int e0   = blockIdx.x * 16;

    if (tid < 16) { s_idx[tid] = senders[e0 + tid]; r_idx[tid] = receivers[e0 + tid]; }
    __syncthreads();

    // Stage X = [nf[s] | nf[r] | ef[e]]  (16 x 384) as bf16
    {
        const int r = tid >> 4, t = tid & 15;
        const int s = s_idx[r], rv = r_idx[r];
        const size_t eb = (size_t)(e0 + r) * DIM;
        for (int q = 0; q < 24; ++q) {
            int c = t * 24 + q;
            float v;
            if      (c < 128) v = nf[(size_t)s  * DIM + c];
            else if (c < 256) v = nf[(size_t)rv * DIM + (c - 128)];
            else              v = ef[eb + (c - 256)];
            x_lds[r * 384 + c] = f2bf(v);
        }
    }
    __syncthreads();

    const int m  = lane & 15;
    const int kb = (lane >> 4) * 8;

    // GEMM1: (16x384) @ (384x128), N-tile = wv
    v8f acc = {};
    for (int kt = 0; kt < 12; ++kt) {
        v16bf a = load_a(x_lds, 384, m, kt * 32, kb);
        v16bf b = load_b(eW1p, kt, wv, lane);
        acc = __builtin_amdgcn_wmma_f32_16x16x32_bf16(false, a, false, b, (short)0, acc, false, false);
    }
    {   // bias + relu -> bf16 LDS
        int n = wv * 16 + (lane & 15);
        float bias = eb1[n];
        int rbase = (lane >> 4) * 8;
        for (int i = 0; i < 8; ++i) {
            float v = acc[i] + bias;
            h_lds[(rbase + i) * 128 + n] = f2bf(v > 0.0f ? v : 0.0f);
        }
    }
    __syncthreads();

    // GEMM2: (16x128) @ (128x128)
    v8f acc2 = {};
    for (int kt = 0; kt < 4; ++kt) {
        v16bf a = load_a(h_lds, 128, m, kt * 32, kb);
        v16bf b = load_b(eW2p, kt, wv, lane);
        acc2 = __builtin_amdgcn_wmma_f32_16x16x32_bf16(false, a, false, b, (short)0, acc2, false, false);
    }
    {
        int n = wv * 16 + (lane & 15);
        float bias = eb2[n];
        int rbase = (lane >> 4) * 8;
        for (int i = 0; i < 8; ++i)
            y_lds[(rbase + i) * 128 + n] = acc2[i] + bias;
    }
    __syncthreads();

    // LayerNorm stats (fp32)
    const int r = tid >> 4, t = tid & 15;
    {
        float s = 0.0f, q = 0.0f;
        for (int cc = 0; cc < 8; ++cc) {
            float v = y_lds[r * 128 + t * 8 + cc];
            s += v; q += v * v;
        }
        red_s[tid] = s; red_q[tid] = q;
    }
    __syncthreads();
    if (t == 0) {
        float S = 0.0f, Q = 0.0f;
        for (int k = 0; k < 16; ++k) { S += red_s[(r << 4) + k]; Q += red_q[(r << 4) + k]; }
        float mu  = S * (1.0f / 128.0f);
        float var = Q * (1.0f / 128.0f) - mu * mu;
        mu_lds[r] = mu;
        rs_lds[r] = rsqrtf(var + LN_EPS);
    }
    __syncthreads();
    {
        float mu = mu_lds[r], rs = rs_lds[r];
        const size_t eb = (size_t)(e0 + r) * DIM;
        float lg = 0.0f;
        for (int cc = 0; cc < 8; ++cc) {
            int c = t * 8 + cc;
            float v = (y_lds[r * 128 + c] - mu) * rs * eg[c] + ebeta[c];
            out_ef[eb + c] = v + ef[eb + c];     // residual output
            lg += v * aW[c];                      // attention logit partial
        }
        red_s[tid] = lg;
    }
    __syncthreads();
    if (t == 0) {
        float z = 0.0f;
        for (int k = 0; k < 16; ++k) z += red_s[(r << 4) + k];
        z += ab[0];
        z = z > 0.0f ? z : 0.2f * z;              // leaky_relu 0.2
        logits[e0 + r] = z;
        atomicMaxFloat(&segmax[r_idx[r]], z);
    }
}

// ---------- segment softmax ----------

__global__ void exp_kernel(const float* __restrict__ logits, const int* __restrict__ receivers,
                           const float* __restrict__ segmax, float* __restrict__ segsum,
                           float* __restrict__ exs)
{
    int e = blockIdx.x * blockDim.x + threadIdx.x;
    if (e >= N_EDGES) return;
    int rv = receivers[e];
    float mx = segmax[rv];
    if (!(mx > -3.0e38f)) mx = 0.0f;              // isfinite guard
    float ex = expf(logits[e] - mx);
    exs[e] = ex;
    atomicAdd(&segsum[rv], ex);
}

// agg[rv] += (new_ef - ef) * att ; one thread per (edge, channel)
__global__ void agg_kernel(const float* __restrict__ out_ef, const float* __restrict__ ef,
                           const float* __restrict__ exs, const float* __restrict__ segsum,
                           const int* __restrict__ receivers, float* __restrict__ agg)
{
    unsigned int idx = blockIdx.x * 256u + threadIdx.x;   // < 76,800,000 exactly
    int e = idx >> 7;
    int c = idx & 127;
    int rv = receivers[e];
    float att = exs[e] / segsum[rv];
    size_t off = (size_t)e * DIM + c;
    float v = (out_ef[off] - ef[off]) * att;
    atomicAdd(&agg[(size_t)rv * DIM + c], v);
}

// ---------- node MLP + LN + residual ----------

__global__ void __launch_bounds__(256)
node_kernel(const float* __restrict__ nf, const float* __restrict__ agg,
            const unsigned int* __restrict__ nW1p, const float* __restrict__ nb1,
            const unsigned int* __restrict__ nW2p, const float* __restrict__ nb2,
            const float* __restrict__ ng, const float* __restrict__ nbeta,
            float* __restrict__ out_nf)
{
    __shared__ __align__(16) unsigned short x_lds[16 * 256];
    __shared__ __align__(16) unsigned short h_lds[16 * 128];
    __shared__ __align__(16) float          y_lds[16 * 128];
    __shared__ float red_s[256];
    __shared__ float red_q[256];
    __shared__ float mu_lds[16], rs_lds[16];

    const int tid  = threadIdx.x;
    const int lane = tid & 31;
    const int wv   = tid >> 5;
    const int n0   = blockIdx.x * 16;

    {   // Stage X = [nf | agg]  (16 x 256)
        const int r = tid >> 4, t = tid & 15;
        const size_t nb = (size_t)(n0 + r) * DIM;
        for (int q = 0; q < 16; ++q) {
            int c = t * 16 + q;
            float v = (c < 128) ? nf[nb + c] : agg[nb + (c - 128)];
            x_lds[r * 256 + c] = f2bf(v);
        }
    }
    __syncthreads();

    const int m  = lane & 15;
    const int kb = (lane >> 4) * 8;

    v8f acc = {};
    for (int kt = 0; kt < 8; ++kt) {
        v16bf a = load_a(x_lds, 256, m, kt * 32, kb);
        v16bf b = load_b(nW1p, kt, wv, lane);
        acc = __builtin_amdgcn_wmma_f32_16x16x32_bf16(false, a, false, b, (short)0, acc, false, false);
    }
    {
        int n = wv * 16 + (lane & 15);
        float bias = nb1[n];
        int rbase = (lane >> 4) * 8;
        for (int i = 0; i < 8; ++i) {
            float v = acc[i] + bias;
            h_lds[(rbase + i) * 128 + n] = f2bf(v > 0.0f ? v : 0.0f);
        }
    }
    __syncthreads();

    v8f acc2 = {};
    for (int kt = 0; kt < 4; ++kt) {
        v16bf a = load_a(h_lds, 128, m, kt * 32, kb);
        v16bf b = load_b(nW2p, kt, wv, lane);
        acc2 = __builtin_amdgcn_wmma_f32_16x16x32_bf16(false, a, false, b, (short)0, acc2, false, false);
    }
    {
        int n = wv * 16 + (lane & 15);
        float bias = nb2[n];
        int rbase = (lane >> 4) * 8;
        for (int i = 0; i < 8; ++i)
            y_lds[(rbase + i) * 128 + n] = acc2[i] + bias;
    }
    __syncthreads();

    const int r = tid >> 4, t = tid & 15;
    {
        float s = 0.0f, q = 0.0f;
        for (int cc = 0; cc < 8; ++cc) {
            float v = y_lds[r * 128 + t * 8 + cc];
            s += v; q += v * v;
        }
        red_s[tid] = s; red_q[tid] = q;
    }
    __syncthreads();
    if (t == 0) {
        float S = 0.0f, Q = 0.0f;
        for (int k = 0; k < 16; ++k) { S += red_s[(r << 4) + k]; Q += red_q[(r << 4) + k]; }
        float mu  = S * (1.0f / 128.0f);
        float var = Q * (1.0f / 128.0f) - mu * mu;
        mu_lds[r] = mu;
        rs_lds[r] = rsqrtf(var + LN_EPS);
    }
    __syncthreads();
    {
        float mu = mu_lds[r], rs = rs_lds[r];
        const size_t nb = (size_t)(n0 + r) * DIM;
        for (int cc = 0; cc < 8; ++cc) {
            int c = t * 8 + cc;
            float v = (y_lds[r * 128 + c] - mu) * rs * ng[c] + nbeta[c];
            out_nf[nb + c] = v + nf[nb + c];     // residual
        }
    }
}

// ---------- launch ----------

extern "C" void kernel_launch(void* const* d_in, const int* in_sizes, int n_in,
                              void* d_out, int out_size, void* d_ws, size_t ws_size,
                              hipStream_t stream) {
    const float* nf       = (const float*)d_in[0];
    const float* ef       = (const float*)d_in[1];
    const int*   senders  = (const int*)  d_in[2];
    const int*   receivers= (const int*)  d_in[3];
    const float* eW1      = (const float*)d_in[4];
    const float* eb1      = (const float*)d_in[5];
    const float* eW2      = (const float*)d_in[6];
    const float* eb2      = (const float*)d_in[7];
    const float* eg       = (const float*)d_in[8];
    const float* ebeta    = (const float*)d_in[9];
    const float* nW1      = (const float*)d_in[10];
    const float* nb1      = (const float*)d_in[11];
    const float* nW2      = (const float*)d_in[12];
    const float* nb2      = (const float*)d_in[13];
    const float* ng       = (const float*)d_in[14];
    const float* nbeta    = (const float*)d_in[15];
    const float* aW       = (const float*)d_in[16];
    const float* ab       = (const float*)d_in[17];

    float* out_nf = (float*)d_out;                       // 50000*128
    float* out_ef = out_nf + (size_t)N_NODES * DIM;      // 600000*128

    // workspace partition (dwords); packed-weight bases are 32B aligned
    float* ws     = (float*)d_ws;
    float* segmax = ws;                                  // 50000
    float* segsum = segmax + N_NODES;                    // 50000
    float* logits = segsum + N_NODES;                    // 600000
    float* exs    = logits + N_EDGES;                    // 600000
    float* agg    = exs + N_EDGES;                       // 6,400,000
    unsigned int* eW1p = (unsigned int*)(agg + (size_t)N_NODES * DIM); // 12*2048
    unsigned int* eW2p = eW1p + 12 * 2048;               // 4*2048
    unsigned int* nW1p = eW2p + 4 * 2048;                // 8*2048
    unsigned int* nW2p = nW1p + 8 * 2048;                // 4*2048

    init_kernel<<<(N_NODES * DIM) / 256, 256, 0, stream>>>(segmax, segsum, agg);
    pack_weights<<<(12 * 2048) / 256, 256, 0, stream>>>(eW1, eW1p, 12);
    pack_weights<<<(4  * 2048) / 256, 256, 0, stream>>>(eW2, eW2p, 4);
    pack_weights<<<(8  * 2048) / 256, 256, 0, stream>>>(nW1, nW1p, 8);
    pack_weights<<<(4  * 2048) / 256, 256, 0, stream>>>(nW2, nW2p, 4);

    edge_kernel<<<N_EDGES / 16, 256, 0, stream>>>(nf, ef, senders, receivers,
                                                  eW1p, eb1, eW2p, eb2, eg, ebeta,
                                                  aW, ab, out_ef, logits, segmax);

    exp_kernel<<<(N_EDGES + 255) / 256, 256, 0, stream>>>(logits, receivers, segmax, segsum, exs);

    agg_kernel<<<(N_EDGES * DIM) / 256, 256, 0, stream>>>(out_ef, ef, exs, segsum, receivers, agg);

    node_kernel<<<N_NODES / 16, 256, 0, stream>>>(nf, agg, nW1p, nb1, nW2p, nb2,
                                                  ng, nbeta, out_nf);
}
